// AttentionModel_88905823027207
// MI455X (gfx1250) — matrix operand order repaired and checked by reading, not compiled, not run on hardware
//
#include <hip/hip_runtime.h>

// ---------- types ----------
typedef __attribute__((ext_vector_type(16))) __bf16 v16bf;
typedef __attribute__((ext_vector_type(8)))  __bf16 v8bf;
typedef __attribute__((ext_vector_type(8)))  float  v8f;

#define QKV_SZ 8388608ull  // 4*16*2048*64 bf16 elements per Q/K/V tensor

// ---------- helpers ----------
__device__ __forceinline__ unsigned short f2bf(float f) {
  unsigned u = __float_as_uint(f);
  u += 0x7FFFu + ((u >> 16) & 1u);        // round-to-nearest-even
  return (unsigned short)(u >> 16);
}

__device__ __forceinline__ v16bf cat16(v8bf lo, v8bf hi) {
  return __builtin_shufflevector(lo, hi, 0, 1, 2, 3, 4, 5, 6, 7,
                                 8, 9, 10, 11, 12, 13, 14, 15);
}

// A-operand 16x32 bf16 fragment per CDNA5 ISA layout:
// lane: m = L&15, half = L>>4; VGPR0-3: K = half*8 + {0..7}; VGPR4-7: K = 16 + half*8 + {0..7}
__device__ __forceinline__ v16bf ld_fragA(const unsigned short* base, int row0, int k0, int ldk) {
  int lane = threadIdx.x & 31;
  const unsigned short* p = base + (row0 + (lane & 15)) * ldk + k0 + ((lane >> 4) << 3);
  return cat16(*(const v8bf*)p, *(const v8bf*)(p + 16));
}

// B-operand 32x16 bf16 fragment read from an [N][K] tile (K-contiguous per column):
// lane: n = L&15, half = L>>4; VGPR0-3: K = half*8 + {0..7}; VGPR4-7: K = 16 + half*8 + {0..7}
__device__ __forceinline__ v16bf ld_fragB(const unsigned short* base, int n0, int k0, int ldk) {
  int lane = threadIdx.x & 31;
  const unsigned short* p = base + (n0 + (lane & 15)) * ldk + k0 + ((lane >> 4) << 3);
  return cat16(*(const v8bf*)p, *(const v8bf*)(p + 16));
}

__device__ __forceinline__ v8f wmma_bf16(v16bf a, v16bf b, v8f c) {
  return __builtin_amdgcn_wmma_f32_16x16x32_bf16(false, a, false, b, (short)0, c, false, false);
}

__device__ __forceinline__ v8f zero8() {
  v8f z;
#pragma unroll
  for (int i = 0; i < 8; ++i) z[i] = 0.0f;
  return z;
}

template <int M>
__device__ __forceinline__ float swz_xor(float x) {
  // ds_swizzle group-of-32: xor_mask=M, and_mask=0x1F (stays inside 16-lane halves for M<16)
  return __int_as_float(__builtin_amdgcn_ds_swizzle(__float_as_int(x), (M << 10) | 0x1F));
}

// Async global->LDS 128-bit copy (gfx1250): bypasses VGPRs, tracked by ASYNCcnt.
// LDS operand = low 32 bits of the flat shared pointer (flat->LDS keeps addr[31:0]).
__device__ __forceinline__ void async_g2l_b128(unsigned short* lds_dst, const void* gsrc) {
  unsigned lds_off = (unsigned)(size_t)lds_dst;
  asm volatile("global_load_async_to_lds_b128 %0, %1, off"
               :
               : "v"(lds_off), "v"(gsrc)
               : "memory");
}
__device__ __forceinline__ void wait_async0() {
  asm volatile("s_wait_asynccnt 0" ::: "memory");
}

// =====================================================================
// Kernel 1: QKV projection  C[8192,3072] = X[8192,1024] * Wqkv[1024,3072]
// Block tile 128x128 (8 waves, 64x32 each), K-step 32.
// Epilogue scatters bf16 into Q/K/V tensors laid out [B][H][S][64].
// =====================================================================
__global__ __launch_bounds__(256) void qkv_gemm(const float* __restrict__ X,
                                                const float* __restrict__ W,
                                                unsigned short* __restrict__ qkv) {
  __shared__ __align__(16) unsigned short lds_a[128 * 40];
  __shared__ __align__(16) unsigned short lds_b[128 * 40];
  const int tid = threadIdx.x;
  const int wave = tid >> 5, lane = tid & 31;
  const int tile_n = blockIdx.x % 24, tile_m = blockIdx.x / 24;
  const int row0 = tile_m * 128, n0 = tile_n * 128;
  const int wm = (wave & 1) * 64, wn = (wave >> 1) * 32;

  v8f acc[4][2];
#pragma unroll
  for (int i = 0; i < 4; ++i)
#pragma unroll
    for (int j = 0; j < 2; ++j) acc[i][j] = zero8();

  for (int k0 = 0; k0 < 1024; k0 += 32) {
    __syncthreads();
    {  // stage X tile 128x32 (f32 -> bf16), row-major, pitch 40
      const float* Ab = X + (size_t)row0 * 1024 + k0;
      const int c = (tid & 7) * 4;
#pragma unroll
      for (int it = 0; it < 4; ++it) {
        int r = (tid >> 3) + it * 32;
        float4 v = *(const float4*)(Ab + (size_t)r * 1024 + c);
        uint2 pk;
        pk.x = (unsigned)f2bf(v.x) | ((unsigned)f2bf(v.y) << 16);
        pk.y = (unsigned)f2bf(v.z) | ((unsigned)f2bf(v.w) << 16);
        *(uint2*)&lds_a[r * 40 + c] = pk;
      }
    }
    {  // stage W tile 32x128 transposed -> lds_b[n][k], pitch 40
      const float* Bb = W + (size_t)k0 * 3072 + n0;
      const int kk = tid >> 3, nn = (tid & 7) * 16;
#pragma unroll
      for (int q = 0; q < 4; ++q) {
        float4 w = *(const float4*)(Bb + (size_t)kk * 3072 + nn + q * 4);
        lds_b[(nn + q * 4 + 0) * 40 + kk] = f2bf(w.x);
        lds_b[(nn + q * 4 + 1) * 40 + kk] = f2bf(w.y);
        lds_b[(nn + q * 4 + 2) * 40 + kk] = f2bf(w.z);
        lds_b[(nn + q * 4 + 3) * 40 + kk] = f2bf(w.w);
      }
    }
    __syncthreads();
    v16bf a[4], b[2];
#pragma unroll
    for (int i = 0; i < 4; ++i) a[i] = ld_fragA(lds_a, wm + i * 16, 0, 40);
#pragma unroll
    for (int j = 0; j < 2; ++j) b[j] = ld_fragB(lds_b, wn + j * 16, 0, 40);
#pragma unroll
    for (int i = 0; i < 4; ++i)
#pragma unroll
      for (int j = 0; j < 2; ++j) acc[i][j] = wmma_bf16(a[i], b[j], acc[i][j]);
  }

  // epilogue: scatter into Q/K/V [B][H][S][64] (bf16)
  const int n = lane & 15, mb = (lane >> 4) * 8;
#pragma unroll
  for (int fi = 0; fi < 4; ++fi)
#pragma unroll
    for (int fj = 0; fj < 2; ++fj)
#pragma unroll
      for (int r = 0; r < 8; ++r) {
        int grow = row0 + wm + fi * 16 + mb + r;   // 0..8191
        int gcol = n0 + wn + fj * 16 + n;          // 0..3071
        int part = gcol >> 10;
        int within = gcol & 1023;
        int h = within >> 6, d = within & 63;
        int b2 = grow >> 11, s = grow & 2047;
        qkv[(size_t)part * QKV_SZ +
            (((size_t)(b2 * 16 + h) * 2048 + s) * 64 + d)] = f2bf(acc[fi][fj][r]);
      }
}

// =====================================================================
// Kernel 2: flash attention over one (b,h, 128-query tile) per block.
// Each wave: 16 queries; online softmax; O accumulated in f32 C-frags.
// K tile staged with async global->LDS (ASYNCcnt); V staged transposed.
// =====================================================================
__global__ __launch_bounds__(256) void attn_fa(const unsigned short* __restrict__ qkv,
                                               unsigned short* __restrict__ out) {
  constexpr int SD = 72;  // LDS pitch (halfwords): 144B rows, 16B-aligned
  __shared__ __align__(16) unsigned short k_tile[64 * SD];      // [key][hd]
  __shared__ __align__(16) unsigned short v_tile[64 * SD];      // [hd][key] (transposed)
  __shared__ __align__(16) unsigned short p_tile[8 * 16 * SD];  // per-wave P scratch
  const int tid = threadIdx.x, wave = tid >> 5, lane = tid & 31;
  const int qt = blockIdx.x & 15, bh = blockIdx.x >> 4;
  const unsigned short* Qb = qkv + (size_t)bh * (2048 * 64);
  const unsigned short* Kb = Qb + QKV_SZ;
  const unsigned short* Vb = Qb + 2 * QKV_SZ;
  const int q0 = qt * 128 + wave * 16;
  const int nn = lane & 15, half = lane >> 4;

  // Q fragments straight from global (already bf16 row-major [s][hd])
  v16bf aq0, aq1;
  {
    const unsigned short* qr = Qb + (size_t)(q0 + nn) * 64 + half * 8;
    aq0 = cat16(*(const v8bf*)qr, *(const v8bf*)(qr + 16));
    aq1 = cat16(*(const v8bf*)(qr + 32), *(const v8bf*)(qr + 48));
  }

  v8f o0 = zero8(), o1 = zero8(), o2 = zero8(), o3 = zero8();
  float mrow[8], lrow[8];
#pragma unroll
  for (int r = 0; r < 8; ++r) { mrow[r] = -1e30f; lrow[r] = 0.0f; }

  unsigned short* pw = &p_tile[wave * 16 * SD];

  for (int k0 = 0; k0 < 2048; k0 += 64) {
    __syncthreads();
    {  // stage K [key][hd] via async global->LDS; V transposed manually
      int r = tid >> 2, c = (tid & 3) * 16;
      const unsigned short* ks = Kb + (size_t)(k0 + r) * 64 + c;
      async_g2l_b128(&k_tile[r * SD + c], ks);
      async_g2l_b128(&k_tile[r * SD + c + 8], ks + 8);
      const unsigned short* vs = Vb + (size_t)(k0 + r) * 64 + c;
      union { uint4 q[2]; unsigned short u[16]; } vv;
      vv.q[0] = *(const uint4*)vs;
      vv.q[1] = *(const uint4*)(vs + 8);
#pragma unroll
      for (int i = 0; i < 16; ++i) v_tile[(c + i) * SD + r] = vv.u[i];
    }
    if (k0 + 64 < 2048) {  // uniform branch: prefetch next tiles
      __builtin_prefetch(Kb + (size_t)(k0 + 64) * 64 + tid * 16, 0, 1);
      __builtin_prefetch(Vb + (size_t)(k0 + 64) * 64 + tid * 16, 0, 1);
    }
    wait_async0();  // K-tile LDS writes complete for this wave
    __syncthreads();

    // scores: S(16x64) = Q(16x64) * K^T ; B-operand reads k_tile[key][hd] directly
    v8f sc[4];
#pragma unroll
    for (int j = 0; j < 4; ++j) {
      sc[j] = zero8();
      sc[j] = wmma_bf16(aq0, ld_fragB(k_tile, j * 16, 0, SD), sc[j]);
      sc[j] = wmma_bf16(aq1, ld_fragB(k_tile, j * 16, 32, SD), sc[j]);
    }

    // online softmax per row (row = half*8 + r in C layout)
#pragma unroll
    for (int r = 0; r < 8; ++r) {
      float mx = -1e30f;
#pragma unroll
      for (int j = 0; j < 4; ++j) {
        sc[j][r] *= 0.125f;  // 1/sqrt(64)
        mx = fmaxf(mx, sc[j][r]);
      }
      mx = fmaxf(mx, swz_xor<1>(mx));
      mx = fmaxf(mx, swz_xor<2>(mx));
      mx = fmaxf(mx, swz_xor<4>(mx));
      mx = fmaxf(mx, swz_xor<8>(mx));
      float mnew = fmaxf(mrow[r], mx);
      float alpha = __expf(mrow[r] - mnew);
      mrow[r] = mnew;
      float ps = 0.0f;
#pragma unroll
      for (int j = 0; j < 4; ++j) {
        float p = __expf(sc[j][r] - mnew);
        sc[j][r] = p;
        ps += p;
      }
      ps += swz_xor<1>(ps);
      ps += swz_xor<2>(ps);
      ps += swz_xor<4>(ps);
      ps += swz_xor<8>(ps);
      lrow[r] = lrow[r] * alpha + ps;
      o0[r] *= alpha; o1[r] *= alpha; o2[r] *= alpha; o3[r] *= alpha;
    }

    // C-layout -> A-frag layout via per-wave LDS round trip (bf16)
#pragma unroll
    for (int j = 0; j < 4; ++j)
#pragma unroll
      for (int r = 0; r < 8; ++r)
        pw[(half * 8 + r) * SD + j * 16 + nn] = f2bf(sc[j][r]);

    v16bf pa0 = ld_fragA(pw, 0, 0, SD);
    v16bf pa1 = ld_fragA(pw, 0, 32, SD);
    // O(16x64) += P(16x64) * V(64x64); B-operand reads v_tile[hd][key]
    o0 = wmma_bf16(pa0, ld_fragB(v_tile, 0, 0, SD), o0);
    o0 = wmma_bf16(pa1, ld_fragB(v_tile, 0, 32, SD), o0);
    o1 = wmma_bf16(pa0, ld_fragB(v_tile, 16, 0, SD), o1);
    o1 = wmma_bf16(pa1, ld_fragB(v_tile, 16, 32, SD), o1);
    o2 = wmma_bf16(pa0, ld_fragB(v_tile, 32, 0, SD), o2);
    o2 = wmma_bf16(pa1, ld_fragB(v_tile, 32, 32, SD), o2);
    o3 = wmma_bf16(pa0, ld_fragB(v_tile, 48, 0, SD), o3);
    o3 = wmma_bf16(pa1, ld_fragB(v_tile, 48, 32, SD), o3);
  }

  // epilogue: write [B][S][H][64] == [B][S][D] contiguous, bf16
  const int b = bh >> 4, h = bh & 15;
#pragma unroll
  for (int r = 0; r < 8; ++r) {
    int s = q0 + half * 8 + r;
    float inv = 1.0f / lrow[r];
    size_t base = ((size_t)(b * 2048 + s) * 16 + h) * 64;
    out[base + 0 + nn]  = f2bf(o0[r] * inv);
    out[base + 16 + nn] = f2bf(o1[r] * inv);
    out[base + 32 + nn] = f2bf(o2[r] * inv);
    out[base + 48 + nn] = f2bf(o3[r] * inv);
  }
}

// =====================================================================
// Kernel 3: output projection  C[8192,1024] = Attn_bf16[8192,1024] * Wout[1024,1024]
// Block tile 128x128 (8 waves, 64x32 each). A staged via async global->LDS.
// =====================================================================
__global__ __launch_bounds__(256) void out_gemm(const unsigned short* __restrict__ A,
                                                const float* __restrict__ W,
                                                float* __restrict__ C) {
  __shared__ __align__(16) unsigned short lds_a[128 * 40];
  __shared__ __align__(16) unsigned short lds_b[128 * 40];
  const int tid = threadIdx.x;
  const int wave = tid >> 5, lane = tid & 31;
  const int tile_n = blockIdx.x & 7, tile_m = blockIdx.x >> 3;
  const int row0 = tile_m * 128, n0 = tile_n * 128;
  const int wm = (wave & 1) * 64, wn = (wave >> 1) * 32;

  v8f acc[4][2];
#pragma unroll
  for (int i = 0; i < 4; ++i)
#pragma unroll
    for (int j = 0; j < 2; ++j) acc[i][j] = zero8();

  for (int k0 = 0; k0 < 1024; k0 += 32) {
    __syncthreads();
    {  // stage A tile 128x32 (already bf16): async 16B copies, no VGPR round trip
      int r = tid >> 1, coff = (tid & 1) * 16;
      const unsigned short* src = A + (size_t)(row0 + r) * 1024 + k0 + coff;
      async_g2l_b128(&lds_a[r * 40 + coff], src);
      async_g2l_b128(&lds_a[r * 40 + coff + 8], src + 8);
    }
    {  // stage Wout tile 32x128 transposed -> lds_b[n][k]
      const float* Bb = W + (size_t)k0 * 1024 + n0;
      const int kk = tid >> 3, mm = (tid & 7) * 16;
#pragma unroll
      for (int q = 0; q < 4; ++q) {
        float4 w = *(const float4*)(Bb + (size_t)kk * 1024 + mm + q * 4);
        lds_b[(mm + q * 4 + 0) * 40 + kk] = f2bf(w.x);
        lds_b[(mm + q * 4 + 1) * 40 + kk] = f2bf(w.y);
        lds_b[(mm + q * 4 + 2) * 40 + kk] = f2bf(w.z);
        lds_b[(mm + q * 4 + 3) * 40 + kk] = f2bf(w.w);
      }
    }
    wait_async0();
    __syncthreads();
    v16bf a[4], b[2];
#pragma unroll
    for (int i = 0; i < 4; ++i) a[i] = ld_fragA(lds_a, wm + i * 16, 0, 40);
#pragma unroll
    for (int j = 0; j < 2; ++j) b[j] = ld_fragB(lds_b, wn + j * 16, 0, 40);
#pragma unroll
    for (int i = 0; i < 4; ++i)
#pragma unroll
      for (int j = 0; j < 2; ++j) acc[i][j] = wmma_bf16(a[i], b[j], acc[i][j]);
  }

  const int n = lane & 15, mb = (lane >> 4) * 8;
#pragma unroll
  for (int fi = 0; fi < 4; ++fi)
#pragma unroll
    for (int fj = 0; fj < 2; ++fj)
#pragma unroll
      for (int r = 0; r < 8; ++r) {
        int grow = row0 + wm + fi * 16 + mb + r;
        int gcol = n0 + wn + fj * 16 + n;
        C[(size_t)grow * 1024 + gcol] = acc[fi][fj][r];
      }
}

// =====================================================================
extern "C" void kernel_launch(void* const* d_in, const int* in_sizes, int n_in,
                              void* d_out, int out_size, void* d_ws, size_t ws_size,
                              hipStream_t stream) {
  (void)in_sizes; (void)n_in; (void)out_size; (void)ws_size;
  const float* x     = (const float*)d_in[0];  // [4,2048,1024]
  const float* w_qkv = (const float*)d_in[1];  // [1024,3072]
  const float* w_out = (const float*)d_in[2];  // [1024,1024]
  unsigned short* qkv  = (unsigned short*)d_ws;          // 3 * 16MB bf16
  unsigned short* attn = qkv + 3 * QKV_SZ;               // 16MB bf16

  qkv_gemm<<<dim3(64 * 24), dim3(256), 0, stream>>>(x, w_qkv, qkv);
  attn_fa<<<dim3(1024), dim3(256), 0, stream>>>(qkv, attn);
  out_gemm<<<dim3(512), dim3(256), 0, stream>>>(attn, w_out, (float*)d_out);
}
